// TopicLayer_19585050869773
// MI455X (gfx1250) — compile-verified, matrix operand order checked by reading, not run on hardware
//
#include <hip/hip_runtime.h>

// Problem constants
#define B_  16
#define L_  1024
#define D_  512
#define T_  32
#define FT_ 128
#define FS_ 128
#define M_TOTAL 4224   // T*FT + FS
#define C_OUT   256    // FT + FS

typedef __attribute__((ext_vector_type(16))) __bf16 v16bf;
typedef __attribute__((ext_vector_type(8)))  float  v8f;

__device__ __forceinline__ unsigned short f32_to_bf16(float f) {
    unsigned int u = __float_as_uint(f);
    unsigned int rnd = 0x7FFFu + ((u >> 16) & 1u);   // round-to-nearest-even
    return (unsigned short)((u + rnd) >> 16);
}
__device__ __forceinline__ float bf16_to_f32(unsigned short h) {
    return __uint_as_float(((unsigned int)h) << 16);
}

// ---------------------------------------------------------------------------
// fp32 -> bf16 (hi, lo) split conversion for embeddings
// ---------------------------------------------------------------------------
__global__ __launch_bounds__(256) void cvt_embed(const float* __restrict__ src,
                                                 unsigned short* __restrict__ hi,
                                                 unsigned short* __restrict__ lo,
                                                 int n4) {
    int i = blockIdx.x * 256 + threadIdx.x;
    if (i >= n4) return;
    float4 v = ((const float4*)src)[i];
    ushort4 h, l;
    h.x = f32_to_bf16(v.x); l.x = f32_to_bf16(v.x - bf16_to_f32(h.x));
    h.y = f32_to_bf16(v.y); l.y = f32_to_bf16(v.y - bf16_to_f32(h.y));
    h.z = f32_to_bf16(v.z); l.z = f32_to_bf16(v.z - bf16_to_f32(h.z));
    h.w = f32_to_bf16(v.w); l.w = f32_to_bf16(v.w - bf16_to_f32(h.w));
    ((ushort4*)hi)[i] = h;
    ((ushort4*)lo)[i] = l;
}

// ---------------------------------------------------------------------------
// fp32 -> bf16 (hi, lo) for concatenated weights: rows [0,4096)=topic_w,
// rows [4096,4224)=shared_w, D=512 contiguous.
// ---------------------------------------------------------------------------
__global__ __launch_bounds__(256) void cvt_weights(const float* __restrict__ topic_w,
                                                   const float* __restrict__ shared_w,
                                                   unsigned short* __restrict__ hi,
                                                   unsigned short* __restrict__ lo,
                                                   int n4) {
    int i = blockIdx.x * 256 + threadIdx.x;
    if (i >= n4) return;
    int e = i << 2;
    int m = e >> 9;          // / 512
    int d = e & 511;
    const float* s = (m < T_ * FT_) ? (topic_w + ((size_t)m << 9) + d)
                                    : (shared_w + ((size_t)(m - T_ * FT_) << 9) + d);
    float4 v = *(const float4*)s;
    ushort4 h, l;
    h.x = f32_to_bf16(v.x); l.x = f32_to_bf16(v.x - bf16_to_f32(h.x));
    h.y = f32_to_bf16(v.y); l.y = f32_to_bf16(v.y - bf16_to_f32(h.y));
    h.z = f32_to_bf16(v.z); l.z = f32_to_bf16(v.z - bf16_to_f32(h.z));
    h.w = f32_to_bf16(v.w); l.w = f32_to_bf16(v.w - bf16_to_f32(h.w));
    ((ushort4*)hi)[i] = h;
    ((ushort4*)lo)[i] = l;
}

// ---------------------------------------------------------------------------
// gfx1250 async copy: 16 bytes global -> LDS, tracked by ASYNCcnt.
// VDST operand holds the LDS byte offset (low 32 bits of the shared pointer,
// per the aperture mapping); VADDR is the 64-bit global address (GV mode).
// ---------------------------------------------------------------------------
__device__ __forceinline__ void async_cp16(unsigned short* lds,
                                           const unsigned short* g) {
    unsigned off = (unsigned)(unsigned long long)lds;
    asm volatile("global_load_async_to_lds_b128 %0, %1, off"
                 :: "v"(off), "v"(g) : "memory");
}

// ---------------------------------------------------------------------------
// WMMA fragment load from an LDS 16x32 bf16 tile (row stride 32 ushorts).
// ISA 7.12.2 16-bit A layout: lane&15 = row, lane>>4 selects K groups
// {0..7,16..23} vs {8..15,24..31}.  Same layout used for B (col-major cols).
// ---------------------------------------------------------------------------
__device__ __forceinline__ v16bf load_frag(const unsigned short* base, int lane) {
    const int r  = lane & 15;
    const int kh = (lane >> 4) << 3;           // 0 or 8
    union { v16bf v; uint4 q[2]; } f;
    f.q[0] = *(const uint4*)(base + r * 32 + kh);        // K = kh .. kh+7
    f.q[1] = *(const uint4*)(base + r * 32 + 16 + kh);   // K = 16+kh .. 16+kh+7
    return f.v;
}

// ---------------------------------------------------------------------------
// Main GEMM: per batch b, C[m,l] = sum_d Wcat[m,d] * E[b,l,d]  (+bias at store)
// Block tile 128(M) x 128(N), K-step 32, bf16 hi/lo split (3 WMMAs / tile),
// double-buffered async global->LDS staging.
// ---------------------------------------------------------------------------
__global__ __launch_bounds__(256) void topic_gemm(
    const unsigned short* __restrict__ Whi,
    const unsigned short* __restrict__ Wlo,
    const unsigned short* __restrict__ Ehi,
    const unsigned short* __restrict__ Elo,
    const float* __restrict__ topic_b,
    const float* __restrict__ shared_b,
    float* __restrict__ out)
{
    // [buf][hi/lo][128 rows x 32 k]  -> 64 KB total
    __shared__ __align__(16) unsigned short sA[2][2][128 * 32];
    __shared__ __align__(16) unsigned short sB[2][2][128 * 32];

    const int tid  = threadIdx.x;
    const int lane = tid & 31;
    const int wave = tid >> 5;
    const int wm   = wave & 3;       // M: 4 waves x 32 rows
    const int wn   = wave >> 2;      // N: 2 waves x 64 cols

    const int l0 = blockIdx.x * 128;
    const int m0 = blockIdx.y * 128;
    const int b  = blockIdx.z;

    v8f acc[2][4];
#pragma unroll
    for (int mi = 0; mi < 2; ++mi)
#pragma unroll
        for (int ni = 0; ni < 4; ++ni)
            acc[mi][ni] = (v8f){0.f, 0.f, 0.f, 0.f, 0.f, 0.f, 0.f, 0.f};

    // Per-thread staging addresses: 2 chunks x 4 planes = 8 async loads/tile.
    const int c0row[2] = { tid >> 2, (tid + 256) >> 2 };
    const int c0kc[2]  = { (tid & 3) << 3, ((tid + 256) & 3) << 3 };

    auto issue_tile = [&](int buf, int kb) {
        const int kOff = kb * 32;
#pragma unroll
        for (int i = 0; i < 2; ++i) {
            int row = c0row[i];
            int kc  = c0kc[i];
            size_t gA = (size_t)(m0 + row) * D_ + kOff + kc;
            size_t gB = ((size_t)b * L_ + (l0 + row)) * D_ + kOff + kc;
            int s = row * 32 + kc;
            async_cp16(&sA[buf][0][s], Whi + gA);
            async_cp16(&sA[buf][1][s], Wlo + gA);
            async_cp16(&sB[buf][0][s], Ehi + gB);
            async_cp16(&sB[buf][1][s], Elo + gB);
        }
    };

    issue_tile(0, 0);

#pragma unroll 2
    for (int kb = 0; kb < D_ / 32; ++kb) {
        const int cur = kb & 1;
        if (kb + 1 < D_ / 32) {
            issue_tile(cur ^ 1, kb + 1);
            // allow the 8 just-issued loads to stay in flight
            asm volatile("s_wait_asynccnt 0x8" ::: "memory");
        } else {
            asm volatile("s_wait_asynccnt 0x0" ::: "memory");
        }
        __syncthreads();

        v16bf bhi[4], blo[4];
#pragma unroll
        for (int ni = 0; ni < 4; ++ni) {
            bhi[ni] = load_frag(&sB[cur][0][(wn * 64 + ni * 16) * 32], lane);
            blo[ni] = load_frag(&sB[cur][1][(wn * 64 + ni * 16) * 32], lane);
        }
#pragma unroll
        for (int mi = 0; mi < 2; ++mi) {
            v16bf ahi = load_frag(&sA[cur][0][(wm * 32 + mi * 16) * 32], lane);
            v16bf alo = load_frag(&sA[cur][1][(wm * 32 + mi * 16) * 32], lane);
#pragma unroll
            for (int ni = 0; ni < 4; ++ni) {
                acc[mi][ni] = __builtin_amdgcn_wmma_f32_16x16x32_bf16(
                    false, ahi, false, bhi[ni], (short)0, acc[mi][ni], false, false);
                acc[mi][ni] = __builtin_amdgcn_wmma_f32_16x16x32_bf16(
                    false, ahi, false, blo[ni], (short)0, acc[mi][ni], false, false);
                acc[mi][ni] = __builtin_amdgcn_wmma_f32_16x16x32_bf16(
                    false, alo, false, bhi[ni], (short)0, acc[mi][ni], false, false);
            }
        }
        __syncthreads();   // all waves done reading 'cur' before it is re-filled
    }

    // Store with bias; C layout: VGPR r -> M = r + 8*(lane>>4), N = lane&15
    const bool isShared = (m0 >= T_ * FT_);           // uniform per block
    const int  tStride  = B_ * C_OUT * L_;            // 4,194,304
#pragma unroll
    for (int mi = 0; mi < 2; ++mi) {
#pragma unroll
        for (int ni = 0; ni < 4; ++ni) {
            int l = l0 + wn * 64 + ni * 16 + (lane & 15);
#pragma unroll
            for (int r = 0; r < 8; ++r) {
                int m = m0 + wm * 32 + mi * 16 + r + ((lane >> 4) << 3);
                float v = acc[mi][ni][r];
                if (!isShared) {
                    v += topic_b[m];
                    int t = m >> 7;
                    int f = m & 127;
                    out[((t * B_ + b) * C_OUT + f) * L_ + l] = v;
                } else {
                    int fs = m - T_ * FT_;
                    v += shared_b[fs];
                    int base = (b * C_OUT + FT_ + fs) * L_ + l;
#pragma unroll 4
                    for (int t = 0; t < T_; ++t)
                        out[base + t * tStride] = v;
                }
            }
        }
    }
}

// ---------------------------------------------------------------------------
extern "C" void kernel_launch(void* const* d_in, const int* in_sizes, int n_in,
                              void* d_out, int out_size, void* d_ws, size_t ws_size,
                              hipStream_t stream) {
    const float* emb      = (const float*)d_in[0];
    const float* topic_w  = (const float*)d_in[1];
    const float* topic_b  = (const float*)d_in[2];
    const float* shared_w = (const float*)d_in[3];
    const float* shared_b = (const float*)d_in[4];
    float* out = (float*)d_out;

    // Workspace layout (bf16 hi/lo planes): ~42.2 MB total
    unsigned short* Ehi = (unsigned short*)d_ws;
    unsigned short* Elo = Ehi + (size_t)B_ * L_ * D_;
    unsigned short* Whi = Elo + (size_t)B_ * L_ * D_;
    unsigned short* Wlo = Whi + (size_t)M_TOTAL * D_;

    int nE4 = B_ * L_ * D_ / 4;                 // 2,097,152 float4s
    cvt_embed<<<nE4 / 256, 256, 0, stream>>>(emb, Ehi, Elo, nE4);

    int nW4 = M_TOTAL * D_ / 4;                 // 540,672 float4s
    cvt_weights<<<(nW4 + 255) / 256, 256, 0, stream>>>(topic_w, shared_w, Whi, Wlo, nW4);

    dim3 grid(L_ / 128, M_TOTAL / 128, B_);     // (8, 33, 16)
    topic_gemm<<<grid, 256, 0, stream>>>(Whi, Wlo, Ehi, Elo, topic_b, shared_b, out);
}
